// TreeModel_11605001634006
// MI455X (gfx1250) — compile-verified, matrix operand order checked by reading, not run on hardware
//
#include <hip/hip_runtime.h>
#include <hip/hip_bf16.h>

#define TREE_DEPTH 18
#define N_TOTAL ((1 << TREE_DEPTH) - 1)   // 262143 nodes
#define HDIM 128

typedef __attribute__((ext_vector_type(16))) _Float16 v16h;
typedef __attribute__((ext_vector_type(4)))  _Float16 v4h;
typedef __attribute__((ext_vector_type(8)))  float    v8f;

// Packed B-fragment buffer layout (fragment = 512 f16 = 32 lanes x 16 elems):
//   frags [  0, 96)  : W_iou  (24 col-tiles x 4 k-steps)
//   frags [ 96,192)  : U_iou  (24 x 4)
//   frags [192,224)  : W_f    ( 8 x 4)
//   frags [224,256)  : U_f    ( 8 x 4)
//   frags [256,260)  : W_out  ( 1 x 4, cols 5..15 zero-padded)
#define PF_W_IOU 0
#define PF_U_IOU 96
#define PF_W_F   192
#define PF_U_F   224
#define PF_W_OUT 256
#define PF_COUNT 260

// ---------------------------------------------------------------------------
// WMMA helpers (CDNA5 wave32, v_wmma_f32_16x16x32_f16)
// 16-bit operand fragment element mapping (ISA 7.12.2):
//   lane: m = lane&15 (row for A / col for B), hi = lane>>4
//   packed element j = 2v+e of v16h:  K = 16*(v>>2) + 8*hi + 2*(v&3) + e
// ---------------------------------------------------------------------------
__device__ __forceinline__ v8f wmma_f16(v16h a, v16h b, v8f c) {
  return __builtin_amdgcn_wmma_f32_16x16x32_f16(
      /*neg_a=*/false, a, /*neg_b=*/false, b,
      /*c_mod=*/(short)0, c, /*reuse_a=*/false, /*reuse_b=*/false);
}

// A fragment from LDS f16, row = m*mul + add, leading dim ld
__device__ __forceinline__ v16h load_a_lds(const _Float16* lds, int lane,
                                           int k0, int mul, int add, int ld) {
  int m = lane & 15, hi = lane >> 4;
  v16h a;
#pragma unroll
  for (int j = 0; j < 16; ++j) {
    int v = j >> 1, e = j & 1;
    int K = k0 + ((v >> 2) << 4) + (hi << 3) + ((v & 3) << 1) + e;
    a[j] = lds[(m * mul + add) * ld + K];
  }
  return a;
}

// Pre-packed B fragment: two coalesced 16B loads per lane.
__device__ __forceinline__ v16h load_b_packed(const _Float16* packed,
                                              int frag, int lane) {
  return ((const v16h*)packed)[frag * 32 + lane];
}

__device__ __forceinline__ float sigm(float x) {
  return 1.f / (1.f + __expf(-x));
}
__device__ __forceinline__ float tanh_fast(float x) {
  x = fminf(20.f, fmaxf(-20.f, x));
  float e = __expf(-2.f * x);
  return (1.f - e) / (1.f + e);
}

// ---------------------------------------------------------------------------
// One-shot weight packer: f32 weights -> f16 WMMA B fragments in workspace.
// ---------------------------------------------------------------------------
__global__ __launch_bounds__(256) void pack_kernel(
    const float* __restrict__ W_iou, const float* __restrict__ U_iou,
    const float* __restrict__ W_f,  const float* __restrict__ U_f,
    const float* __restrict__ W_out, _Float16* __restrict__ packed)
{
  const int f = blockIdx.x;
  const float* W; int ldw, cb, ks, nvalid;
  if (f < PF_U_IOU)      { int g = f;            W = W_iou; ldw = 384; cb = (g >> 2) * 16; ks = g & 3; nvalid = 16; }
  else if (f < PF_W_F)   { int g = f - PF_U_IOU; W = U_iou; ldw = 384; cb = (g >> 2) * 16; ks = g & 3; nvalid = 16; }
  else if (f < PF_U_F)   { int g = f - PF_W_F;   W = W_f;   ldw = 128; cb = (g >> 2) * 16; ks = g & 3; nvalid = 16; }
  else if (f < PF_W_OUT) { int g = f - PF_U_F;   W = U_f;   ldw = 128; cb = (g >> 2) * 16; ks = g & 3; nvalid = 16; }
  else                   { int g = f - PF_W_OUT; W = W_out; ldw = 5;   cb = 0;             ks = g;     nvalid = 5;  }
  const int k0 = ks * 32;
  for (int e = threadIdx.x; e < 512; e += 256) {
    int lane = e >> 4, j = e & 15;
    int n = lane & 15, hi = lane >> 4;
    int v = j >> 1, eo = j & 1;
    int K = k0 + ((v >> 2) << 4) + (hi << 3) + ((v & 3) << 1) + eo;
    float val = (n < nvalid) ? W[K * ldw + cb + n] : 0.f;
    packed[(size_t)f * 512 + e] = (_Float16)val;
  }
}

// ---------------------------------------------------------------------------
// Per-level fused kernel: embedding gather + input GEMMs + recurrent GEMMs +
// LSTM gating for a tile of 16 nodes per block (8 wave32s).
// ---------------------------------------------------------------------------
__global__ __launch_bounds__(256) void level_kernel(
    const int* __restrict__ x_ids, const int* __restrict__ mask,
    const float* __restrict__ emb_table,
    const float* __restrict__ b_iou, const float* __restrict__ b_f,
    const _Float16* __restrict__ packed,
    _Float16* __restrict__ h_buf, float* __restrict__ c_buf,
    int s0, int nlev, int internal)
{
  __shared__ __align__(16) _Float16 emb_s[16 * HDIM];    // embeddings, f16
  __shared__ __align__(16) _Float16 hch_s[32 * HDIM];    // child h rows, f16
  __shared__ __align__(16) _Float16 hsum_s[16 * HDIM];   // hL + hR, f16
  __shared__ float    iou_s[16 * 384];                   // iou pre-activations
  __shared__ float    f_s[2 * 16 * HDIM];                // forget pre-act (L,R)

  const int tid  = threadIdx.x;
  const int lane = tid & 31;
  const int wave = tid >> 5;
  const int j0   = blockIdx.x * 16;
  const long long cbase = 2LL * (s0 + j0) + 1;  // first child of tile row 0

  // ---- async-stage the 32 contiguous child h rows (8 KB, always in range
  //      because the tree is complete): CDNA5 global->LDS async path ----
  if (internal) {
    const char* src = (const char*)(h_buf + (size_t)cbase * HDIM);
    unsigned lds_base = (unsigned)(uintptr_t)(&hch_s[0]);
    unsigned off = (unsigned)tid * 32u;                 // 32 B per thread
    unsigned lds_addr = lds_base + off;
    unsigned long long gaddr = (unsigned long long)(uintptr_t)(src + off);
    asm volatile(
        "global_load_async_to_lds_b128 %0, %1, off\n\t"
        "global_load_async_to_lds_b128 %0, %1, off offset:16"
        :: "v"(lds_addr), "v"(gaddr) : "memory");
  }

  // ---- stage embeddings (gather), 4 elems at a time ----
  for (int q = tid; q < 16 * (HDIM / 4); q += 256) {
    int j  = q >> 5;             // 32 float4 per 128-wide row
    int k4 = (q & 31) * 4;
    int jj = (j0 + j < nlev) ? (j0 + j) : (nlev - 1);
    long long node = s0 + jj;
    long long row  = (long long)x_ids[node] * (long long)mask[node];
    float4 fv = ((const float4*)(emb_table + row * HDIM))[q & 31];
    v4h hv = {(_Float16)fv.x, (_Float16)fv.y, (_Float16)fv.z, (_Float16)fv.w};
    *(v4h*)(&emb_s[j * HDIM + k4]) = hv;
  }

  if (internal) {
    asm volatile("s_wait_asynccnt 0" ::: "memory");
  }
  __syncthreads();

  if (internal) {
    for (int q = tid; q < 16 * (HDIM / 4); q += 256) {
      int j = q >> 5, k4 = (q & 31) * 4;
      v4h a = *(const v4h*)(&hch_s[(2 * j) * HDIM + k4]);
      v4h b = *(const v4h*)(&hch_s[(2 * j + 1) * HDIM + k4]);
      v4h s;
#pragma unroll
      for (int i = 0; i < 4; ++i) s[i] = (_Float16)((float)a[i] + (float)b[i]);
      *(v4h*)(&hsum_s[j * HDIM + k4]) = s;
    }
    __syncthreads();
  }

  // ---- WMMA phase: 24 iou tiles (+16 forget tiles if internal) ----
  const int ntiles = internal ? 40 : 24;
  for (int t = wave; t < ntiles; t += 8) {
    v8f acc = {};
    const int n = lane & 15, hi = lane >> 4;
    if (t < 24) {                                   // iou columns [16t,16t+16)
      int cb = t * 16;
#pragma unroll
      for (int ks = 0; ks < 4; ++ks) {
        v16h a = load_a_lds(emb_s, lane, ks * 32, 1, 0, HDIM);
        v16h b = load_b_packed(packed, PF_W_IOU + t * 4 + ks, lane);
        acc = wmma_f16(a, b, acc);
      }
      if (internal) {
#pragma unroll
        for (int ks = 0; ks < 4; ++ks) {
          v16h a = load_a_lds(hsum_s, lane, ks * 32, 1, 0, HDIM);
          v16h b = load_b_packed(packed, PF_U_IOU + t * 4 + ks, lane);
          acc = wmma_f16(a, b, acc);
        }
      }
#pragma unroll
      for (int r = 0; r < 8; ++r)
        iou_s[(r + 8 * hi) * 384 + cb + n] = acc[r];
    } else {                                        // forget pre-act, child ch
      int tt = t - 24;
      int ch = tt >> 3;                             // 0 = left, 1 = right
      int ct = tt & 7;
      int cb = ct * 16;
#pragma unroll
      for (int ks = 0; ks < 4; ++ks) {
        v16h a = load_a_lds(emb_s, lane, ks * 32, 1, 0, HDIM);
        v16h b = load_b_packed(packed, PF_W_F + ct * 4 + ks, lane);
        acc = wmma_f16(a, b, acc);
      }
#pragma unroll
      for (int ks = 0; ks < 4; ++ks) {
        v16h a = load_a_lds(hch_s, lane, ks * 32, 2, ch, HDIM);  // rows 2m+ch
        v16h b = load_b_packed(packed, PF_U_F + ct * 4 + ks, lane);
        acc = wmma_f16(a, b, acc);
      }
#pragma unroll
      for (int r = 0; r < 8; ++r)
        f_s[(ch * 16 + r + 8 * hi) * HDIM + cb + n] = acc[r];
    }
  }
  __syncthreads();

  // ---- elementwise LSTM gating ----
  for (int idx = tid; idx < 16 * HDIM; idx += 256) {
    int j = idx >> 7, k = idx & 127;
    if (j0 + j < nlev) {
      long long node = s0 + j0 + j;
      float ig = iou_s[j * 384 + k]       + b_iou[k];
      float og = iou_s[j * 384 + 128 + k] + b_iou[128 + k];
      float ug = iou_s[j * 384 + 256 + k] + b_iou[256 + k];
      float fc = 0.f;
      if (internal) {
        long long cl = cbase + 2 * j;
        float fl = sigm(f_s[(0 * 16 + j) * HDIM + k] + b_f[k]);
        float fr = sigm(f_s[(1 * 16 + j) * HDIM + k] + b_f[k]);
        fc = fl * c_buf[cl * HDIM + k] + fr * c_buf[(cl + 1) * HDIM + k];
      }
      float cv = sigm(ig) * tanh_fast(ug) + fc;
      float hv = sigm(og) * tanh_fast(cv);
      c_buf[node * HDIM + k] = cv;
      h_buf[node * HDIM + k] = (_Float16)hv;
    }
  }
}

// ---------------------------------------------------------------------------
// Output head: out = h_all @ W_out + b_out  (128 -> 5), W_out zero-padded to
// 16 columns inside the packed B fragment. One wave per 16-node tile.
// ---------------------------------------------------------------------------
__global__ __launch_bounds__(256) void out_kernel(
    const _Float16* __restrict__ h_buf, const _Float16* __restrict__ packed,
    const float* __restrict__ b_out, float* __restrict__ out)
{
  const int lane = threadIdx.x & 31;
  const int wave = threadIdx.x >> 5;
  const long long base = ((long long)blockIdx.x * 8 + wave) * 16;
  const int m = lane & 15, hi = lane >> 4;
  long long row = base + m;
  if (row > (long long)(N_TOTAL - 1)) row = N_TOTAL - 1;

  v8f acc = {};
#pragma unroll
  for (int ks = 0; ks < 4; ++ks) {
    int k0 = ks * 32;
    v16h a;
#pragma unroll
    for (int j = 0; j < 16; ++j) {
      int v = j >> 1, e = j & 1;
      int K = k0 + ((v >> 2) << 4) + (hi << 3) + ((v & 3) << 1) + e;
      a[j] = h_buf[row * HDIM + K];
    }
    v16h b = load_b_packed(packed, PF_W_OUT + ks, lane);
    acc = wmma_f16(a, b, acc);
  }
  const int n = lane & 15;
  if (n < 5) {
#pragma unroll
    for (int r = 0; r < 8; ++r) {
      long long mr = base + r + 8 * hi;
      if (mr < N_TOTAL) out[mr * 5 + n] = acc[r] + b_out[n];
    }
  }
}

// ---------------------------------------------------------------------------
extern "C" void kernel_launch(void* const* d_in, const int* in_sizes, int n_in,
                              void* d_out, int out_size, void* d_ws, size_t ws_size,
                              hipStream_t stream) {
  const int*   x_ids     = (const int*)d_in[0];
  const int*   mask      = (const int*)d_in[1];
  const float* emb_table = (const float*)d_in[2];
  const float* W_iou     = (const float*)d_in[3];
  const float* U_iou     = (const float*)d_in[4];
  const float* b_iou     = (const float*)d_in[5];
  const float* W_f       = (const float*)d_in[6];
  const float* U_f       = (const float*)d_in[7];
  const float* b_f       = (const float*)d_in[8];
  const float* W_out     = (const float*)d_in[9];
  const float* b_out     = (const float*)d_in[10];

  // workspace: c (f32, N*128) | h (f16, N*128) | packed weights (f16)
  char* ws = (char*)d_ws;
  float*    c_buf  = (float*)ws;
  _Float16* h_buf  = (_Float16*)(ws + (size_t)N_TOTAL * HDIM * sizeof(float));
  _Float16* packed = (_Float16*)(ws + (size_t)N_TOTAL * HDIM * sizeof(float)
                                    + (size_t)N_TOTAL * HDIM * sizeof(_Float16));

  // pack weights into WMMA B-fragment layout (L2-resident, ~260 KB)
  pack_kernel<<<PF_COUNT, 256, 0, stream>>>(W_iou, U_iou, W_f, U_f, W_out, packed);

  // leaves -> root; stream ordering carries the level dependency
  for (int d = TREE_DEPTH - 1; d >= 0; --d) {
    int nlev = 1 << d;
    int s0   = nlev - 1;
    int blocks = (nlev + 15) / 16;
    level_kernel<<<blocks, 256, 0, stream>>>(
        x_ids, mask, emb_table, b_iou, b_f, packed,
        h_buf, c_buf, s0, nlev, (d < TREE_DEPTH - 1) ? 1 : 0);
  }

  // output head: 16384 tiles of 16 nodes, 8 waves per block
  int tiles  = (N_TOTAL + 15) / 16;           // 16384
  int blocks = (tiles + 7) / 8;               // 2048
  out_kernel<<<blocks, 256, 0, stream>>>(h_buf, packed, b_out, (float*)d_out);
}
